// EnhancedCPCEncoder_39204461478353
// MI455X (gfx1250) — matrix operand order repaired and checked by reading, not compiled, not run on hardware
//
#include <hip/hip_runtime.h>
#include <hip/hip_fp16.h>

// ---------------- types ----------------
typedef __attribute__((ext_vector_type(16))) _Float16 v16h;
typedef __attribute__((ext_vector_type(8)))  float    v8f;
typedef __attribute__((ext_vector_type(4)))  int      v4i;
typedef __attribute__((address_space(1))) v4i* gv4i_p;  // global int4*
typedef __attribute__((address_space(3))) v4i* lv4i_p;  // LDS int4*

// ---------------- problem constants ----------------
#define B_   128
#define T0_  4096
#define T1_  2048
#define T2_  1024
#define T3_  512
#define C1_  32
#define C2_  64
#define C3_  128
#define HID_ 256
#define G3_  768            // 3*HID
#define M_   (B_*T3_)       // 65536 GEMM rows

// ---------------- math helpers ----------------
__device__ __forceinline__ float gelu_tanh(float x) {
  float x3 = x * x * x;
  return 0.5f * x * (1.0f + tanhf(0.7978845608028654f * (x + 0.044715f * x3)));
}
__device__ __forceinline__ float sigmoidf_(float x) { return 1.0f / (1.0f + expf(-x)); }

// ---------------- WMMA fragment helpers (ISA 7.12.2 layouts, wave32) ----------------
__device__ __forceinline__ v16h load_a_frag(const _Float16* p, int ld, int lane) {
  int row = lane & 15;
  int kb  = (lane & 16) ? 8 : 0;
  const _Float16* q = p + (long)row * ld + kb;
  v16h a;
#pragma unroll
  for (int e = 0; e < 8; ++e) a[e] = q[e];
#pragma unroll
  for (int e = 0; e < 8; ++e) a[8 + e] = q[16 + e];
  return a;
}
// B fragment from pre-packed layout: lane's 16 halves are contiguous (32B aligned).
__device__ __forceinline__ v16h load_b_packed(const _Float16* frag, int lane) {
  return *(const v16h*)(frag + lane * 16);
}
__device__ __forceinline__ v8f wmma_f16(v16h a, v16h b, v8f c) {
  return __builtin_amdgcn_wmma_f32_16x16x32_f16(false, a, false, b, (short)0, c, false, false);
}

// ---------------- kernels ----------------
__global__ void k_zero_u32(unsigned* p, int n) {
  int i = blockIdx.x * blockDim.x + threadIdx.x;
  if (i < n) p[i] = 0u;
}

// Pack f32 weight W[K x N] (row-major) into f16 B-fragment layout:
// out[((ntile*(K/32)+kc)*32 + lane)*16 + e] = W[kc*32 + ((lane&16)?8:0) + (e&7) + ((e&8)?16:0)][ntile*16 + (lane&15)]
__global__ void k_pack_b(const float* __restrict__ w, _Float16* __restrict__ out,
                         int K, int N) {
  int idx = blockIdx.x * blockDim.x + threadIdx.x;
  if (idx >= K * N) return;
  int e = idx & 15;
  int l = (idx >> 4) & 31;
  int frag = idx >> 9;                 // ntile*(K/32)+kc
  int kchunks = K >> 5;
  int ntile = frag / kchunks, kc = frag - ntile * kchunks;
  int k = kc * 32 + ((l & 16) ? 8 : 0) + (e & 7) + ((e & 8) ? 16 : 0);
  int n = ntile * 16 + (l & 15);
  out[idx] = (_Float16)w[(long)k * N + n];
}

// conv0: [B,4096] (x * 1e20) -> [B,2048,32], fused RMSNorm(ch) + gelu.
__global__ void k_conv0(const float* __restrict__ x, const float* __restrict__ w,
                        const float* __restrict__ bb, const float* __restrict__ scale,
                        float* __restrict__ out) {
  int lane = threadIdx.x & 31, wid = threadIdx.x >> 5;
  int pos = blockIdx.x * 8 + wid;
  int bi = pos / T1_, to = pos % T1_;
  int c = lane;
  float acc = bb[c];
#pragma unroll
  for (int k = 0; k < 9; ++k) {
    int ti = 2 * to + k - 3;           // SAME pad: lo=3, hi=4
    float xv = (ti >= 0 && ti < T0_) ? x[bi * T0_ + ti] * 1e20f : 0.0f;
    acc += xv * w[k * C1_ + c];
  }
  float ms = acc * acc;
#pragma unroll
  for (int off = 16; off > 0; off >>= 1) ms += __shfl_xor(ms, off, 32);
  float h = acc * rsqrtf(ms * (1.0f / C1_) + 1e-6f) * scale[c];
  out[(long)pos * C1_ + c] = gelu_tanh(h);
}

// conv1: [B,2048,32] -> [B,1024,64]. 256 thr = 4 pos x 64 ch.
__global__ void k_conv1(const float* __restrict__ a0, const float* __restrict__ w,
                        const float* __restrict__ bb, const float* __restrict__ scale,
                        float* __restrict__ out) {
  __shared__ float part[8];
  int tid = threadIdx.x;
  int c = tid & 63, p = tid >> 6;
  int lane = tid & 31, wid = tid >> 5;
  int pos = blockIdx.x * 4 + p;
  int bi = pos / T2_, to = pos % T2_;
  float acc = bb[c];
  for (int k = 0; k < 9; ++k) {
    int ti = 2 * to + k - 3;
    if (ti >= 0 && ti < T1_) {
      const float* ap = a0 + (long)(bi * T1_ + ti) * C1_;
      const float* wp = w + k * C1_ * C2_ + c;
#pragma unroll
      for (int ci = 0; ci < C1_; ++ci) acc += ap[ci] * wp[ci * C2_];
    }
  }
  float ms = acc * acc;
#pragma unroll
  for (int off = 16; off > 0; off >>= 1) ms += __shfl_xor(ms, off, 32);
  if (lane == 0) part[wid] = ms;
  __syncthreads();
  float tot = part[2 * p] + part[2 * p + 1];
  float h = acc * rsqrtf(tot * (1.0f / C2_) + 1e-6f) * scale[c];
  out[(long)pos * C2_ + c] = gelu_tanh(h);
}

// conv2: [B,1024,64] -> [B,512,128] (f16 out). 256 thr = 2 pos x 128 ch.
__global__ void k_conv2(const float* __restrict__ a1, const float* __restrict__ w,
                        const float* __restrict__ bb, const float* __restrict__ scale,
                        _Float16* __restrict__ out) {
  __shared__ float part[8];
  int tid = threadIdx.x;
  int c = tid & 127, p = tid >> 7;
  int lane = tid & 31, wid = tid >> 5;
  int pos = blockIdx.x * 2 + p;
  int bi = pos / T3_, to = pos % T3_;
  float acc = bb[c];
  for (int k = 0; k < 9; ++k) {
    int ti = 2 * to + k - 3;
    if (ti >= 0 && ti < T2_) {
      const float* ap = a1 + (long)(bi * T2_ + ti) * C2_;
      const float* wp = w + k * C2_ * C3_ + c;
#pragma unroll
      for (int ci = 0; ci < C2_; ++ci) acc += ap[ci] * wp[ci * C3_];
    }
  }
  float ms = acc * acc;
#pragma unroll
  for (int off = 16; off > 0; off >>= 1) ms += __shfl_xor(ms, off, 32);
  if (lane == 0) part[wid] = ms;
  __syncthreads();
  float tot = part[4 * p] + part[4 * p + 1] + part[4 * p + 2] + part[4 * p + 3];
  float h = acc * rsqrtf(tot * (1.0f / C3_) + 1e-6f) * scale[c];
  out[(long)pos * C3_ + c] = (_Float16)gelu_tanh(h);
}

// weight-normalized projection: pvn[:,n] = v[:,n]/(||v[:,n]||+1e-8)*g[n]  (f32 out, packed later)
__global__ void k_projw(const float* __restrict__ v, const float* __restrict__ g,
                        float* __restrict__ pvn) {
  __shared__ float red[256];
  int n = blockIdx.x, r = threadIdx.x;
  float val = v[(long)r * 256 + n];
  red[r] = val * val;
  __syncthreads();
  for (int s = 128; s > 0; s >>= 1) { if (r < s) red[r] += red[r + s]; __syncthreads(); }
  float norm = sqrtf(red[0]);
  pvn[(long)r * 256 + n] = val / (norm + 1e-8f) * g[n];
}

// xi = A2[65536x128] @ WI[128x768] + bi  -> f32 (WMMA f16, f32 acc). 8 waves/block.
__global__ void k_xi(const _Float16* __restrict__ a2, const _Float16* __restrict__ wip,
                     const float* __restrict__ bi, float* __restrict__ xi) {
  int lane = threadIdx.x & 31, wave = threadIdx.x >> 5;
  int tile = blockIdx.x * 8 + wave;    // 4096 x 48 tiles
  int mt = tile / 48, nt = tile % 48;
  int m0 = mt * 16, n0 = nt * 16;
  int col = lane & 15, rowbase = (lane & 16) ? 8 : 0;
  float bias = bi[n0 + col];
  v8f acc;
#pragma unroll
  for (int r = 0; r < 8; ++r) acc[r] = bias;
#pragma unroll
  for (int kc = 0; kc < 4; ++kc) {
    v16h a = load_a_frag(a2 + (long)m0 * C3_ + kc * 32, C3_, lane);
    v16h b = load_b_packed(wip + (long)(nt * 4 + kc) * 512, lane);
    acc = wmma_f16(a, b, acc);
  }
#pragma unroll
  for (int r = 0; r < 8; ++r)
    __builtin_nontemporal_store(acc[r], &xi[(long)(m0 + rowbase + r) * G3_ + n0 + col]);
}

// GRU recurrence: persistent grid (64 blocks x 2 waves = 128 tiles over [128x256]).
// Recurrent weights staged in LDS once (48KB/WG) via async-to-LDS, 24 WMMAs/tile/step.
__global__ void k_scan(const float* __restrict__ xi, const _Float16* __restrict__ whp,
                       const float* __restrict__ bhn, _Float16* __restrict__ hseq,
                       unsigned* __restrict__ sync) {
  __shared__ _Float16 swh[6 * 8 * 512];          // 6 ntile-chunks x 8 kc x 512 halves = 48KB
  int lane = threadIdx.x & 31, wave = threadIdx.x >> 5;
  int id = blockIdx.x * 2 + wave;                // 0..127
  int mt = id >> 4, nt = id & 15;
  int m0 = mt * 16, n0 = nt * 16;
  int nt0 = (blockIdx.x & 7) * 2;                // block's two nts: nt0, nt0+1
  int col = lane & 15, rowbase = (lane & 16) ? 8 : 0;
  float hb = bhn[n0 + col];
  const int HLD = T3_ * HID_;                    // 131072: batch-row stride in hseq
  const int CH = 8 * 512;                        // halves per ntile-chunk

  // ---- stage this block's 6 packed-wh ntile chunks into LDS (ASYNCcnt path) ----
#if __has_builtin(__builtin_amdgcn_global_load_async_to_lds_b128)
  for (int u = threadIdx.x; u < 6 * 512; u += blockDim.x) {   // 16B units
    int chunk = u >> 9, off = u & 511;
    int g = chunk >> 1, w = chunk & 1;
    const _Float16* gp = whp + (long)((g * 16 + nt0 + w) * 8) * 512 + off * 8;
    _Float16* lp = &swh[chunk * CH + off * 8];
    __builtin_amdgcn_global_load_async_to_lds_b128(
        (gv4i_p)(_Float16*)gp, (lv4i_p)lp, 0, 0);
  }
#if __has_builtin(__builtin_amdgcn_s_wait_asynccnt)
  __builtin_amdgcn_s_wait_asynccnt(0);
#endif
#else
  for (int u = threadIdx.x; u < 6 * 256; u += blockDim.x) {   // 32B units
    int chunk = u >> 8, off = u & 255;
    int g = chunk >> 1, w = chunk & 1;
    *(v16h*)&swh[chunk * CH + off * 16] =
        *(const v16h*)(whp + (long)((g * 16 + nt0 + w) * 8) * 512 + off * 16);
  }
#endif
  __syncthreads();

  const _Float16* myB0 = &swh[(0 * 2 + wave) * CH];
  const _Float16* myB1 = &swh[(1 * 2 + wave) * CH];
  const _Float16* myB2 = &swh[(2 * 2 + wave) * CH];

  for (int t = 0; t < T3_; ++t) {
    v8f acc0 = {}, acc1 = {}, acc2 = {};
    const _Float16* abase = hseq + (long)m0 * HLD + (long)(t - 1) * HID_;
#pragma unroll
    for (int kc = 0; kc < 8; ++kc) {
      v16h a = {};
      if (t != 0) a = load_a_frag(abase + kc * 32, HLD, lane);
      acc0 = wmma_f16(a, load_b_packed(myB0 + kc * 512, lane), acc0);
      acc1 = wmma_f16(a, load_b_packed(myB1 + kc * 512, lane), acc1);
      acc2 = wmma_f16(a, load_b_packed(myB2 + kc * 512, lane), acc2);
    }
#pragma unroll
    for (int r = 0; r < 8; ++r) {
      int b = m0 + rowbase + r;
      long xb = (long)b * T3_ * G3_ + (long)t * G3_;
      float xr = __builtin_nontemporal_load(&xi[xb + 0 * HID_ + n0 + col]);
      float xz = __builtin_nontemporal_load(&xi[xb + 1 * HID_ + n0 + col]);
      float xn = __builtin_nontemporal_load(&xi[xb + 2 * HID_ + n0 + col]);
      float hp = (t == 0) ? 0.0f
                          : (float)hseq[(long)b * HLD + (long)(t - 1) * HID_ + n0 + col];
      float rr = sigmoidf_(xr + acc0[r]);
      float zz = sigmoidf_(xz + acc1[r]);
      float nn = tanhf(xn + rr * (acc2[r] + hb));
      float hv = (1.0f - zz) * nn + zz * hp;
      hseq[(long)b * HLD + (long)t * HID_ + n0 + col] = (_Float16)hv;
    }
    // grid-wide barrier: monotonically increasing counter (reset each launch)
    __syncthreads();
    if (threadIdx.x == 0) {
      __threadfence();
      atomicAdd(sync, 1u);
      unsigned target = (unsigned)gridDim.x * (unsigned)(t + 1);
      while (atomicAdd(sync, 0u) < target) __builtin_amdgcn_s_sleep(8);
    }
    __syncthreads();
  }
}

// y = hseq[65536x256] @ Wp[256x256] + proj_b   (f32 out into d_out)
__global__ void k_proj(const _Float16* __restrict__ hseq, const _Float16* __restrict__ wpp,
                       const float* __restrict__ pb, float* __restrict__ out) {
  int lane = threadIdx.x & 31, wave = threadIdx.x >> 5;
  int tile = blockIdx.x * 8 + wave;    // 4096 x 16 tiles
  int mt = tile >> 4, nt = tile & 15;
  int m0 = mt * 16, n0 = nt * 16;
  int col = lane & 15, rowbase = (lane & 16) ? 8 : 0;
  float bias = pb[n0 + col];
  v8f acc;
#pragma unroll
  for (int r = 0; r < 8; ++r) acc[r] = bias;
#pragma unroll
  for (int kc = 0; kc < 8; ++kc) {
    v16h a = load_a_frag(hseq + (long)m0 * HID_ + kc * 32, HID_, lane);
    v16h b = load_b_packed(wpp + (long)(nt * 8 + kc) * 512, lane);
    acc = wmma_f16(a, b, acc);
  }
#pragma unroll
  for (int r = 0; r < 8; ++r)
    out[(long)(m0 + rowbase + r) * HID_ + n0 + col] = acc[r];
}

// in-place row L2 normalize: y /= (||y||+1e-8) if ||y|| > 1e-6
__global__ void k_norm(float* __restrict__ y) {
  __shared__ float red[256];
  long row = blockIdx.x;
  int tid = threadIdx.x;
  float v = y[row * 256 + tid];
  red[tid] = v * v;
  __syncthreads();
  for (int s = 128; s > 0; s >>= 1) { if (tid < s) red[tid] += red[tid + s]; __syncthreads(); }
  float norm = sqrtf(red[0]);
  if (norm > 1e-6f) y[row * 256 + tid] = v / (norm + 1e-8f);
}

// ---------------- launch ----------------
extern "C" void kernel_launch(void* const* d_in, const int* in_sizes, int n_in,
                              void* d_out, int out_size, void* d_ws, size_t ws_size,
                              hipStream_t stream) {
  const float* x   = (const float*)d_in[0];
  const float* w0  = (const float*)d_in[1];
  const float* b0  = (const float*)d_in[2];
  const float* s0  = (const float*)d_in[3];
  const float* w1  = (const float*)d_in[4];
  const float* b1  = (const float*)d_in[5];
  const float* s1  = (const float*)d_in[6];
  const float* w2  = (const float*)d_in[7];
  const float* b2  = (const float*)d_in[8];
  const float* s2  = (const float*)d_in[9];
  const float* gwi = (const float*)d_in[10];
  const float* gbi = (const float*)d_in[11];
  const float* gwh = (const float*)d_in[12];
  const float* gbh = (const float*)d_in[13];
  const float* pv  = (const float*)d_in[14];
  const float* pg  = (const float*)d_in[15];
  const float* pb  = (const float*)d_in[16];
  float* out = (float*)d_out;

  char* ws = (char*)d_ws;
  float*    a0  = (float*)(ws + 0);                //  33,554,432 B
  float*    a1  = (float*)(ws + 33554432ull);      //  33,554,432 B
  _Float16* a2  = (_Float16*)(ws + 67108864ull);   //  16,777,216 B
  _Float16* wip = (_Float16*)(ws + 83886080ull);   //     196,608 B  (packed wi)
  _Float16* whp = (_Float16*)(ws + 84082688ull);   //     393,216 B  (packed wh)
  _Float16* wpp = (_Float16*)(ws + 84475904ull);   //     131,072 B  (packed proj w)
  float*    pvn = (float*)(ws + 84606976ull);      //     262,144 B  (normalized proj w, f32)
  float*    xi  = (float*)(ws + 84869120ull);      // 201,326,592 B
  _Float16* hs  = (_Float16*)(ws + 286195712ull);  //  33,554,432 B
  unsigned* sy  = (unsigned*)(ws + 319750144ull);  //          64 B

  k_zero_u32<<<1, 64, 0, stream>>>(sy, 16);
  k_conv0<<<B_ * T1_ / 8, 256, 0, stream>>>(x, w0, b0, s0, a0);
  k_conv1<<<B_ * T2_ / 4, 256, 0, stream>>>(a0, w1, b1, s1, a1);
  k_conv2<<<B_ * T3_ / 2, 256, 0, stream>>>(a1, w2, b2, s2, a2);
  k_pack_b<<<(C3_ * G3_ + 255) / 256, 256, 0, stream>>>(gwi, wip, C3_, G3_);
  k_pack_b<<<(HID_ * G3_ + 255) / 256, 256, 0, stream>>>(gwh, whp, HID_, G3_);
  k_projw<<<256, 256, 0, stream>>>(pv, pg, pvn);
  k_pack_b<<<(HID_ * HID_ + 255) / 256, 256, 0, stream>>>(pvn, wpp, HID_, HID_);
  k_xi<<<(4096 * 48) / 8, 256, 0, stream>>>(a2, wip, gbi, xi);
  k_scan<<<64, 64, 0, stream>>>(xi, whp, gbh, hs, sy);
  k_proj<<<(4096 * 16) / 8, 256, 0, stream>>>(hs, wpp, pb, out);
  k_norm<<<M_, 256, 0, stream>>>(out);
}
